// InfoConsistency_37194416783500
// MI455X (gfx1250) — matrix operand order repaired and checked by reading, compiled, not run on hardware
//
#include <hip/hip_runtime.h>
#include <hip/hip_bf16.h>
#include <math.h>

typedef __attribute__((ext_vector_type(16))) _Float16 v16h;
typedef __attribute__((ext_vector_type(8)))  float    v8f;

// Static problem configuration (matches reference setup_inputs()):
constexpr int   B_        = 8;
constexpr int   CF        = 32;   // feature channels
constexpr int   CP        = 4;    // prob channels
constexpr int   S_        = 64;   // spatial size
constexpr int   L_        = 32;   // overlap length (axis = 0 -> slice D dim)
constexpr float LAMBDA_KL = 0.3f;
constexpr float CLAMP_EPS = 1e-6f;
constexpr float COS_EPS   = 1e-8f;

constexpr int S2   = S_ * S_;        // 4096
constexpr int S3   = S_ * S_ * S_;   // 262144
constexpr int NCOL = L_ * S2;        // 131072 columns per batch

// cos kernel tiling
constexpr int TILES_PER_BATCH  = NCOL / 16;                 // 8192
constexpr int TPW              = 32;                        // tiles per wave
constexpr int WAVES_PER_BATCH  = TILES_PER_BATCH / TPW;     // 256
constexpr int WAVES_PER_BLOCK  = 8;                         // 256 threads, wave32
constexpr int BLOCKS_PER_BATCH = WAVES_PER_BATCH / WAVES_PER_BLOCK; // 32
constexpr int COS_BLOCKS       = B_ * BLOCKS_PER_BATCH;     // 256

// kl kernel tiling
constexpr int KL_ELEMS_PER_BATCH  = CP * NCOL;              // 524288
constexpr int KL_BLOCKS_PER_BATCH = 64;
constexpr int KL_THREADS          = 256;
constexpr int KL_ELEMS_PER_BLOCK  = KL_ELEMS_PER_BATCH / KL_BLOCKS_PER_BATCH; // 8192
constexpr int KL_ELEMS_PER_THREAD = KL_ELEMS_PER_BLOCK / KL_THREADS;          // 32
constexpr int KL_BLOCKS           = B_ * KL_BLOCKS_PER_BATCH;                 // 512

// ---------------------------------------------------------------------------
// Cosine term: per 16-column tile, 3 WMMAs give diag(S^T T), diag(S^T S),
// diag(T^T T) over the K=32 channel dimension in one shot each.
// ---------------------------------------------------------------------------
__global__ __launch_bounds__(256) void cos_wmma_kernel(
    const float* __restrict__ sf, const float* __restrict__ tf,
    const int* __restrict__ cur_start, const int* __restrict__ adj_start,
    float* __restrict__ cosPartials)
{
    const int lane        = threadIdx.x & 31;
    const int waveInBlock = threadIdx.x >> 5;
    const int b           = blockIdx.x / BLOCKS_PER_BATCH;
    const int waveInBatch = (blockIdx.x % BLOCKS_PER_BATCH) * WAVES_PER_BLOCK + waveInBlock;

    const int cs = cur_start[b];
    const int as = adj_start[b];
    const float* __restrict__ sfb = sf + (size_t)b * CF * S3;
    const float* __restrict__ tfb = tf + (size_t)b * CF * S3;

    const int m      = lane & 15;   // A-row / B-col / tile column index
    const int laneHi = lane >> 4;

    float acc = 0.0f;

    for (int t = 0; t < TPW; ++t) {
        const int tile = waveInBatch * TPW + t;
        const int col  = tile * 16 + m;          // flat (d,h,w) index in slice
        const int d    = col >> 12;              // / S2
        const int rem  = col & (S2 - 1);
        const int offS = (cs + d) * S2 + rem;    // spatial offset, cur slice
        const int offT = (as + d) * S2 + rem;    // spatial offset, adj slice

        v16h sA, sB, tA, tB;
#pragma unroll
        for (int h = 0; h < 16; ++h) {
            // A-matrix 16x32 f16 layout: lanes0-15 halves map K {0-7,16-23},
            // lanes16-31 map K {8-15,24-31}.
            const int kA = ((h < 8) ? h : (h + 8)) + 8 * laneHi;
            // B-matrix 32x16 f16 layout: same K order as A.
            const int kB = ((h < 8) ? h : (h + 8)) + 8 * laneHi;
            sA[h] = (_Float16)sfb[(size_t)kA * S3 + offS];
            sB[h] = (_Float16)sfb[(size_t)kB * S3 + offS];
            tA[h] = (_Float16)tfb[(size_t)kA * S3 + offT];
            tB[h] = (_Float16)tfb[(size_t)kB * S3 + offT];
        }

        const v8f zero = {};
        v8f dotv = __builtin_amdgcn_wmma_f32_16x16x32_f16(false, sA, false, tB,
                                                          (short)0, zero, false, false);
        v8f nsv  = __builtin_amdgcn_wmma_f32_16x16x32_f16(false, sA, false, sB,
                                                          (short)0, zero, false, false);
        v8f ntv  = __builtin_amdgcn_wmma_f32_16x16x32_f16(false, tA, false, tB,
                                                          (short)0, zero, false, false);

        // Diagonal of D (f32 16x16): VGPR j, lanes0-15 -> M=j,N=lane;
        // lanes16-31 -> M=j+8,N=lane-16.  Diag lives in lanes 0-7 (j=lane)
        // and lanes 24-31 (j=lane-24).
#pragma unroll
        for (int j = 0; j < 8; ++j) {
            const bool mine = (lane == j) || (lane == j + 24);
            const float den = fmaxf(sqrtf(nsv[j]), COS_EPS) *
                              fmaxf(sqrtf(ntv[j]), COS_EPS);
            const float r = dotv[j] / den;
            acc += mine ? r : 0.0f;
        }
    }

    // deterministic wave reduction (wave32)
#pragma unroll
    for (int off = 16; off > 0; off >>= 1)
        acc += __shfl_xor(acc, off, 32);

    __shared__ float smem[WAVES_PER_BLOCK];
    if (lane == 0) smem[waveInBlock] = acc;
    __syncthreads();
    if (threadIdx.x == 0) {
        float s = 0.0f;
#pragma unroll
        for (int w = 0; w < WAVES_PER_BLOCK; ++w) s += smem[w];
        cosPartials[blockIdx.x] = s;
    }
}

// ---------------------------------------------------------------------------
// KL term: sum over slice of tp * (log(tp) - log(sp)) with clamping.
// ---------------------------------------------------------------------------
__global__ __launch_bounds__(KL_THREADS) void kl_kernel(
    const float* __restrict__ sp, const float* __restrict__ tp,
    const int* __restrict__ cur_start, const int* __restrict__ adj_start,
    float* __restrict__ klPartials)
{
    const int b   = blockIdx.x / KL_BLOCKS_PER_BATCH;
    const int blk = blockIdx.x % KL_BLOCKS_PER_BATCH;
    const int cs  = cur_start[b];
    const int as  = adj_start[b];
    const float* __restrict__ spb = sp + (size_t)b * CP * S3;
    const float* __restrict__ tpb = tp + (size_t)b * CP * S3;

    const int base = blk * KL_ELEMS_PER_BLOCK;
    float acc = 0.0f;
#pragma unroll 4
    for (int i = 0; i < KL_ELEMS_PER_THREAD; ++i) {
        const int e   = base + i * KL_THREADS + (int)threadIdx.x;
        const int c   = e / NCOL;
        const int rem = e - c * NCOL;
        const int d   = rem >> 12;            // / S2
        const int r2  = rem & (S2 - 1);
        float sv = spb[(size_t)c * S3 + (cs + d) * S2 + r2];
        float tv = tpb[(size_t)c * S3 + (as + d) * S2 + r2];
        sv = fminf(fmaxf(sv, CLAMP_EPS), 1.0f - CLAMP_EPS);
        tv = fminf(fmaxf(tv, CLAMP_EPS), 1.0f - CLAMP_EPS);
        acc += tv * (logf(tv) - logf(sv));
    }

#pragma unroll
    for (int off = 16; off > 0; off >>= 1)
        acc += __shfl_xor(acc, off, 32);

    __shared__ float smem[KL_THREADS / 32];
    const int lane = threadIdx.x & 31;
    const int wv   = threadIdx.x >> 5;
    if (lane == 0) smem[wv] = acc;
    __syncthreads();
    if (threadIdx.x == 0) {
        float s = 0.0f;
#pragma unroll
        for (int w = 0; w < KL_THREADS / 32; ++w) s += smem[w];
        klPartials[blockIdx.x] = s;
    }
}

// ---------------------------------------------------------------------------
// Final fixed-order combine: loss_b = (1 - cosSum_b/N) + 0.3*klSum_b; mean_b.
// ---------------------------------------------------------------------------
__global__ void final_kernel(const float* __restrict__ cosPartials,
                             const float* __restrict__ klPartials,
                             float* __restrict__ out)
{
    __shared__ float lossSm[B_];
    const int t = threadIdx.x;
    if (t < B_) {
        float cosSum = 0.0f;
        for (int i = 0; i < BLOCKS_PER_BATCH; ++i)
            cosSum += cosPartials[t * BLOCKS_PER_BATCH + i];
        float klSum = 0.0f;
        for (int i = 0; i < KL_BLOCKS_PER_BATCH; ++i)
            klSum += klPartials[t * KL_BLOCKS_PER_BATCH + i];
        const float cosLoss = 1.0f - cosSum / (float)NCOL;
        lossSm[t] = cosLoss + LAMBDA_KL * klSum;
    }
    __syncthreads();
    if (t == 0) {
        float s = 0.0f;
        for (int i = 0; i < B_; ++i) s += lossSm[i];
        out[0] = s / (float)B_;
    }
}

extern "C" void kernel_launch(void* const* d_in, const int* in_sizes, int n_in,
                              void* d_out, int out_size, void* d_ws, size_t ws_size,
                              hipStream_t stream) {
    const float* sf = (const float*)d_in[0];
    const float* tf = (const float*)d_in[1];
    const float* sp = (const float*)d_in[2];
    const float* tp = (const float*)d_in[3];
    const int*   cs = (const int*)d_in[4];
    const int*   as = (const int*)d_in[5];
    // d_in[6..8] = axis / overlap_len / full_size (static in this config)

    float* ws          = (float*)d_ws;
    float* cosPartials = ws;                     // COS_BLOCKS floats
    float* klPartials  = ws + COS_BLOCKS;        // KL_BLOCKS floats
    // All partials are written unconditionally each call -> no clearing needed.

    cos_wmma_kernel<<<COS_BLOCKS, 256, 0, stream>>>(sf, tf, cs, as, cosPartials);
    kl_kernel<<<KL_BLOCKS, KL_THREADS, 0, stream>>>(sp, tp, cs, as, klPartials);
    final_kernel<<<1, 32, 0, stream>>>(cosPartials, klPartials, (float*)d_out);
}
